// Ins_48438641164491
// MI455X (gfx1250) — matrix operand order repaired and checked by reading, not compiled, not run on hardware
//
#include <hip/hip_runtime.h>
#include <hip/hip_bf16.h>
#include <float.h>

typedef __attribute__((ext_vector_type(16))) _Float16 v16h;
typedef __attribute__((ext_vector_type(8)))  _Float16 v8h;
typedef __attribute__((ext_vector_type(4)))  _Float16 v4h;
typedef __attribute__((ext_vector_type(8)))  float    v8f;

#define NB1   256          // phase-1 blocks
#define T1    256          // threads per block
#define NCAND (NB1 * 8)    // 2048 candidates per list for the finalize block
#define NINS  8

__device__ __forceinline__ bool kgt(float v, int i, float v2, int i2) {
  // "greater" with top_k tie-break: equal values -> lower index wins
  return (v > v2) || (v == v2 && i < i2);
}

__device__ __forceinline__ void kins8(float v, int i, float* av, int* ai) {
  if (!kgt(v, i, av[7], ai[7])) return;
  av[7] = v; ai[7] = i;
  #pragma unroll
  for (int p = 7; p > 0; --p) {
    if (kgt(av[p], ai[p], av[p-1], ai[p-1])) {
      float tv = av[p]; av[p] = av[p-1]; av[p-1] = tv;
      int tx = ai[p];  ai[p] = ai[p-1]; ai[p-1] = tx;
    }
  }
}

// ---------------------------------------------------------------------------
// Phase 1: grid-stride scan of A[:,0,bag]; per-thread register top8/bot8,
// then block-level LDS argmax selection; writes 8 top + 8 bottom (negated)
// candidates per block to workspace.
// ---------------------------------------------------------------------------
__global__ void topk_phase1(const float2* __restrict__ A2, const int* __restrict__ bagp, int N,
                            float* __restrict__ topv, int* __restrict__ topi,
                            float* __restrict__ botv, int* __restrict__ boti) {
  __shared__ float cv[T1 * 8];
  __shared__ int   ci[T1 * 8];
  __shared__ float rv[T1];
  __shared__ int   ri[T1];
  __shared__ int   rp[T1];

  const int tid = threadIdx.x;
  const int bag = bagp[0];

  float tv[8]; int ti[8]; float bv[8]; int bi[8];
  #pragma unroll
  for (int j = 0; j < 8; ++j) {
    tv[j] = -FLT_MAX; ti[j] = 0x7fffffff;
    bv[j] = -FLT_MAX; bi[j] = 0x7fffffff;
  }

  const int stride = gridDim.x * blockDim.x;
  for (int i = blockIdx.x * blockDim.x + tid; i < N; i += stride) {
    float2 p = A2[i];                 // coalesced 8B load of both columns
    float v = bag ? p.y : p.x;
    kins8(v,  i, tv, ti);             // top candidates
    kins8(-v, i, bv, bi);             // bottom candidates (max of negated)
  }

  for (int pass = 0; pass < 2; ++pass) {
    __syncthreads();
    #pragma unroll
    for (int j = 0; j < 8; ++j) {
      cv[tid * 8 + j] = pass ? bv[j] : tv[j];
      ci[tid * 8 + j] = pass ? bi[j] : ti[j];
    }
    __syncthreads();
    for (int r = 0; r < 8; ++r) {
      float bvv = -FLT_MAX; int bii = 0x7fffffff; int bpp = 0;
      for (int j = tid; j < T1 * 8; j += T1)
        if (kgt(cv[j], ci[j], bvv, bii)) { bvv = cv[j]; bii = ci[j]; bpp = j; }
      rv[tid] = bvv; ri[tid] = bii; rp[tid] = bpp;
      __syncthreads();
      for (int s = T1 / 2; s > 0; s >>= 1) {
        if (tid < s && kgt(rv[tid + s], ri[tid + s], rv[tid], ri[tid])) {
          rv[tid] = rv[tid + s]; ri[tid] = ri[tid + s]; rp[tid] = rp[tid + s];
        }
        __syncthreads();
      }
      if (tid == 0) {
        if (pass == 0) { topv[blockIdx.x * 8 + r] = rv[0]; topi[blockIdx.x * 8 + r] = ri[0]; }
        else           { botv[blockIdx.x * 8 + r] = rv[0]; boti[blockIdx.x * 8 + r] = ri[0]; }
        cv[rp[0]] = -FLT_MAX;   // remove winner for next round
      }
      __syncthreads();
    }
  }
}

// ---------------------------------------------------------------------------
// Phase 2 (single block): final 8-of-NCAND selection for both lists, gather
// the 16 rows of h into LDS as f16, run the 16x512 @ 512x16(C=2 padded)
// GEMM on wave 0 with v_wmma_f32_16x16x32_f16 (fully unrolled, b128 LDS
// fragment loads), then softmax + labels.
// ---------------------------------------------------------------------------
__global__ void finalize_kernel(const float* __restrict__ h,
                                const float* __restrict__ W,
                                const float* __restrict__ bvec,
                                const float* __restrict__ topv, const int* __restrict__ topi,
                                const float* __restrict__ botv, const int* __restrict__ boti,
                                float* __restrict__ out) {
  __shared__ float cv[NCAND];
  __shared__ int   ci[NCAND];
  __shared__ float rv[256];
  __shared__ int   ri[256];
  __shared__ int   rp[256];
  __shared__ int   idxsh[16];
  __shared__ _Float16 Ah[16 * 512];   // gathered instance rows, f16 (row-major)
  __shared__ _Float16 Wt[16 * 512];   // padded B transposed: Wt[n*512+k]
  __shared__ float Dsh[16 * 16];      // WMMA result dump

  const int tid = threadIdx.x;

  // ---- final top-8 / bottom-8 selection ----
  for (int pass = 0; pass < 2; ++pass) {
    __syncthreads();
    for (int j = tid; j < NCAND; j += 256) {
      cv[j] = pass ? botv[j] : topv[j];
      ci[j] = pass ? boti[j] : topi[j];
    }
    __syncthreads();
    for (int r = 0; r < 8; ++r) {
      float bvv = -FLT_MAX; int bii = 0x7fffffff; int bpp = 0;
      for (int j = tid; j < NCAND; j += 256)
        if (kgt(cv[j], ci[j], bvv, bii)) { bvv = cv[j]; bii = ci[j]; bpp = j; }
      rv[tid] = bvv; ri[tid] = bii; rp[tid] = bpp;
      __syncthreads();
      for (int s = 128; s > 0; s >>= 1) {
        if (tid < s && kgt(rv[tid + s], ri[tid + s], rv[tid], ri[tid])) {
          rv[tid] = rv[tid + s]; ri[tid] = ri[tid + s]; rp[tid] = rp[tid + s];
        }
        __syncthreads();
      }
      if (tid == 0) {
        idxsh[pass * 8 + r] = ri[0];
        cv[rp[0]] = -FLT_MAX;
      }
      __syncthreads();
    }
  }

  // ---- stage padded/transposed W: Wt[n][k] = (n<2) ? W[k][n] : 0 ----
  for (int j = tid; j < 16 * 512; j += 256) {
    int n = j >> 9, k = j & 511;
    Wt[j] = (n < 2) ? (_Float16)W[k * 2 + n] : (_Float16)0.0f;
  }
  // ---- gather 16 rows of h (float4 loads, 4-half LDS stores) ----
  for (int f = tid; f < 16 * 128; f += 256) {
    int r = f >> 7, c4 = f & 127;
    float4 x = ((const float4*)(h + (size_t)idxsh[r] * 512))[c4];
    v4h y = { (_Float16)x.x, (_Float16)x.y, (_Float16)x.z, (_Float16)x.w };
    *(v4h*)&Ah[r * 512 + c4 * 4] = y;
  }
  __syncthreads();

  // ---- GEMM on wave 0: D(16x16) = A(16x512) x B(512x16), K = 16 steps of 32
  if (tid < 32) {
    const int hi = tid >> 4;          // lane half
    const int mn = tid & 15;          // A row == B/D column for this lane
    const _Float16* arow = &Ah[mn * 512];
    const _Float16* brow = &Wt[mn * 512];
    v8f acc = {};
    #pragma unroll
    for (int t = 0; t < 16; ++t) {
      // A fragment: elems 0..7 -> K = t*32+hi*8.. ; elems 8..15 -> +16
      v8h alo = *(const v8h*)&arow[t * 32 + hi * 8];
      v8h ahi = *(const v8h*)&arow[t * 32 + 16 + hi * 8];
      // B fragment: elems 0..7 -> K = t*32+hi*8.. ; elems 8..15 -> +16
      v8h blo = *(const v8h*)&brow[t * 32 + hi * 8];
      v8h bhi = *(const v8h*)&brow[t * 32 + 16 + hi * 8];
      v16h a  = __builtin_shufflevector(alo, ahi, 0,1,2,3,4,5,6,7,8,9,10,11,12,13,14,15);
      v16h bm = __builtin_shufflevector(blo, bhi, 0,1,2,3,4,5,6,7,8,9,10,11,12,13,14,15);
      acc = __builtin_amdgcn_wmma_f32_16x16x32_f16(
          /*neg_a=*/false, a, /*neg_b=*/false, bm,
          /*c_mod=*/(short)0, acc, /*reuse_a=*/false, /*reuse_b=*/false);
    }
    #pragma unroll
    for (int r2 = 0; r2 < 8; ++r2) Dsh[(hi * 8 + r2) * 16 + mn] = acc[r2];
  }
  __syncthreads();

  // ---- softmax over 2 classes + labels ----
  if (tid < 16) {
    float l0 = Dsh[tid * 16 + 0] + bvec[0];
    float l1 = Dsh[tid * 16 + 1] + bvec[1];
    float mx = fmaxf(l0, l1);
    float e0 = expf(l0 - mx), e1 = expf(l1 - mx);
    float s = e0 + e1;
    out[tid] = (tid < NINS) ? 1.0f : 0.0f;        // ins_labels
    out[16 + tid * 2 + 0] = e0 / s;               // ins_logits
    out[16 + tid * 2 + 1] = e1 / s;
  }
}

extern "C" void kernel_launch(void* const* d_in, const int* in_sizes, int n_in,
                              void* d_out, int out_size, void* d_ws, size_t ws_size,
                              hipStream_t stream) {
  const float*  h   = (const float*)d_in[0];    // (N, 512) f32
  const float2* A2  = (const float2*)d_in[1];   // (N, 1, 2) f32 viewed as float2
  const float*  W   = (const float*)d_in[2];    // (512, 2) f32
  const float*  b   = (const float*)d_in[3];    // (2,) f32
  const int*    bag = (const int*)d_in[4];      // scalar int
  const int N = in_sizes[1] / 2;
  float* out = (float*)d_out;

  char* ws = (char*)d_ws;
  float* topv = (float*)ws; ws += NCAND * sizeof(float);
  int*   topi = (int*)ws;   ws += NCAND * sizeof(int);
  float* botv = (float*)ws; ws += NCAND * sizeof(float);
  int*   boti = (int*)ws;   ws += NCAND * sizeof(int);

  topk_phase1<<<NB1, T1, 0, stream>>>(A2, bag, N, topv, topi, botv, boti);
  finalize_kernel<<<1, 256, 0, stream>>>(h, W, b, topv, topi, botv, boti, out);
}
